// BDHBlock_12352325943547
// MI455X (gfx1250) — compile-verified, matrix operand order checked by reading, not run on hardware
//
#include <hip/hip_runtime.h>

// Problem constants (reference: B=2, S=2048, H=1024, NH=16, D=64, FF=4096)
#define BSZ   2
#define SEQ   2048
#define HID   1024
#define NHEAD 16
#define DHEAD 64
#define FFD   4096
#define MDIM  (BSZ * SEQ)   // 4096 token rows

typedef __attribute__((ext_vector_type(16))) __bf16 v16bf;
typedef __attribute__((ext_vector_type(8)))  float  v8f;
typedef unsigned short u16b;

// ---------- CDNA5 async global->LDS copy path (inline asm) -------------------
// GLOBAL_LOAD_ASYNC_TO_LDS_B128: per-lane 16B copy memory->LDS, ASYNCcnt-
// tracked, no VGPR round trip.  vdst = VGPR with LDS byte address (generic
// LDS pointer low 32 bits per flat-aperture rule), vaddr = 64-bit global addr.
#define USE_ASYNC_ASM 1

__device__ __forceinline__ void cp16(u16b* l, const u16b* g) {
#if USE_ASYNC_ASM
    asm volatile("global_load_async_to_lds_b128 %0, %1, off"
                 :: "v"((unsigned)(unsigned long long)l), "v"(g)
                 : "memory");
#else
    uint4 t = *(const uint4*)g;
    *(uint4*)l = t;
#endif
}

__device__ __forceinline__ void async_wait0() {
#if USE_ASYNC_ASM
    asm volatile("s_wait_asynccnt 0" ::: "memory");
#endif
}

// ---------- helpers ----------------------------------------------------------

__device__ __forceinline__ u16b f2bf(float f) {
    unsigned int u = __float_as_uint(f);
    u += 0x7fffu + ((u >> 16) & 1u);          // round-to-nearest-even
    return (u16b)(u >> 16);
}

__device__ __forceinline__ v8f zero8() {
    v8f z = {0.f, 0.f, 0.f, 0.f, 0.f, 0.f, 0.f, 0.f};
    return z;
}

union FragU { v16bf v; unsigned int u[8]; };

__device__ __forceinline__ v8f wmma_bf16(v16bf a, v16bf b, v8f c) {
    return __builtin_amdgcn_wmma_f32_16x16x32_bf16(
        /*neg_a=*/false, a, /*neg_b=*/false, b,
        /*c_mod=*/(short)0, c, /*reuse_a=*/false, /*reuse_b=*/false);
}

// A-fragment (16x32 bf16, MxK) or B-fragment in NT form (B^T row-major (N,K)).
// ISA 7.12.2: lane holds row M = lane&15; hi-half lanes get K+8; VGPR d covers
// K pair {k0,k0+1} with k0 = (d>=4?16:0) + 2*(d&3).  t is row-major, ld even.
__device__ __forceinline__ v16bf frag_nt(const u16b* t, int row, int ld, int kbase) {
    FragU r;
    const unsigned int* t32 = (const unsigned int*)t;
    const int lane = threadIdx.x & 31;
    const int base = (row + (lane & 15)) * ld + kbase + ((lane >> 4) << 3);
#pragma unroll
    for (int d = 0; d < 8; ++d) {
        int k0 = ((d & 4) << 2) + ((d & 3) << 1);
        r.u[d] = t32[(base + k0) >> 1];
    }
    return r.v;
}

// B-fragment in NN form: B[k][n] with t row-major (K x N).  lane holds column
// n = col + lane&15; gather K pairs down the column and pack into dwords.
__device__ __forceinline__ v16bf frag_nn(const u16b* t, int col, int ld, int kbase) {
    FragU r;
    const int lane = threadIdx.x & 31;
    const int n = col + (lane & 15);
    const int kb = kbase + ((lane >> 4) << 3);
#pragma unroll
    for (int d = 0; d < 8; ++d) {
        int k0 = kb + ((d & 4) << 2) + ((d & 3) << 1);
        unsigned int lo = t[k0 * ld + n];
        unsigned int h2 = t[(k0 + 1) * ld + n];
        r.u[d] = lo | (h2 << 16);
    }
    return r.v;
}

// ---------- elementwise kernels ---------------------------------------------

__global__ void cvt_kernel(const float* __restrict__ w, u16b* __restrict__ o, int n) {
    int i = blockIdx.x * blockDim.x + threadIdx.x;
    if (i < n) o[i] = f2bf(w[i]);
}

__global__ void cvt_mask_kernel(const float* __restrict__ w, const float* __restrict__ m,
                                u16b* __restrict__ o, int n) {
    int i = blockIdx.x * blockDim.x + threadIdx.x;
    if (i < n) o[i] = f2bf(w[i] * m[i]);
}

// LayerNorm over H=1024, one row per block (256 threads, 8 wave32).
__global__ __launch_bounds__(256) void ln_kernel(
    const float* __restrict__ x, const float* __restrict__ g,
    const float* __restrict__ b, u16b* __restrict__ out)
{
    __shared__ float ss[8], ss2[8];
    const int row = blockIdx.x;
    const int tid = threadIdx.x;
    const float* xr = x + (size_t)row * HID;
    float lv[4], s = 0.f, s2 = 0.f;
#pragma unroll
    for (int i = 0; i < 4; ++i) {
        lv[i] = xr[tid + i * 256];
        s += lv[i];
        s2 += lv[i] * lv[i];
    }
#pragma unroll
    for (int m = 1; m < 32; m <<= 1) {
        s  += __shfl_xor(s,  m, 32);
        s2 += __shfl_xor(s2, m, 32);
    }
    if ((tid & 31) == 0) { ss[tid >> 5] = s; ss2[tid >> 5] = s2; }
    __syncthreads();
    s = 0.f; s2 = 0.f;
#pragma unroll
    for (int w = 0; w < 8; ++w) { s += ss[w]; s2 += ss2[w]; }
    const float mu   = s * (1.f / HID);
    const float var  = s2 * (1.f / HID) - mu * mu;
    const float rstd = rsqrtf(var + 1e-5f);
#pragma unroll
    for (int i = 0; i < 4; ++i) {
        int c = tid + i * 256;
        out[(size_t)row * HID + c] = f2bf((lv[i] - mu) * rstd * g[c] + b[c]);
    }
}

// ---------- generic WMMA GEMM:  C[M,N] = act(A[M,K] @ W[N,K]^T + bias) (+res)
// flags: bit0 = bf16 output, bit1 = ReLU, bit2 = add f32 residual
// 128x128 tile, K-tile 64, double-buffered LDS with async global->LDS staging.
__global__ __launch_bounds__(256) void gemm_bf16_kernel(
    const u16b* __restrict__ A, const u16b* __restrict__ W,
    const float* __restrict__ bias, const float* __restrict__ res,
    void* __restrict__ out, int M, int N, int K, int flags)
{
    __shared__ __align__(16) u16b sA[2][128 * 64];   // 2 x 16 KB
    __shared__ __align__(16) u16b sW[2][128 * 64];   // 2 x 16 KB

    const int tid  = threadIdx.x;
    const int wid  = tid >> 5;
    const int lane = tid & 31;
    const int hi   = lane >> 4;
    const int nl   = lane & 15;
    const int wm   = wid & 3;          // 4 row groups x 32 rows
    const int wn   = wid >> 2;         // 2 col groups x 64 cols
    const int m0   = blockIdx.y * 128;
    const int n0   = blockIdx.x * 128;

    // staging coords: 4 x 16B chunks per array per thread (128x64 bf16 tile)
    int rr[4], cc[4];
#pragma unroll
    for (int i = 0; i < 4; ++i) {
        int flat = tid * 4 + i;        // 0..1023
        rr[i] = flat >> 3;             // 0..127
        cc[i] = (flat & 7) << 3;       // 0..56 (elements, 16B chunks)
    }

    auto stage = [&](int bi, int k0) {
#pragma unroll
        for (int i = 0; i < 4; ++i)
            cp16(&sA[bi][rr[i] * 64 + cc[i]], A + (size_t)(m0 + rr[i]) * K + k0 + cc[i]);
#pragma unroll
        for (int i = 0; i < 4; ++i)
            cp16(&sW[bi][rr[i] * 64 + cc[i]], W + (size_t)(n0 + rr[i]) * K + k0 + cc[i]);
    };

    v8f acc[2][4];
#pragma unroll
    for (int mi = 0; mi < 2; ++mi)
#pragma unroll
        for (int ni = 0; ni < 4; ++ni) acc[mi][ni] = zero8();

    stage(0, 0);
    int buf = 0;
    for (int k0 = 0; k0 < K; k0 += 64) {
        async_wait0();                 // current buffer's async copies done
        __syncthreads();               // visible to all waves; prev reads done
        if (k0 + 64 < K) stage(buf ^ 1, k0 + 64);   // overlap next tile
#pragma unroll
        for (int kc = 0; kc < 2; ++kc) {
            v16bf af[2], bfr[4];
#pragma unroll
            for (int mi = 0; mi < 2; ++mi)
                af[mi] = frag_nt(sA[buf], wm * 32 + mi * 16, 64, kc * 32);
#pragma unroll
            for (int ni = 0; ni < 4; ++ni)
                bfr[ni] = frag_nt(sW[buf], wn * 64 + ni * 16, 64, kc * 32);
#pragma unroll
            for (int mi = 0; mi < 2; ++mi)
#pragma unroll
                for (int ni = 0; ni < 4; ++ni)
                    acc[mi][ni] = wmma_bf16(af[mi], bfr[ni], acc[mi][ni]);
        }
        buf ^= 1;
    }

    // fused epilogue: bias (+relu) (+residual), f32 or bf16 store
#pragma unroll
    for (int mi = 0; mi < 2; ++mi) {
#pragma unroll
        for (int ni = 0; ni < 4; ++ni) {
            const int gn = n0 + wn * 64 + ni * 16 + nl;
            const float bv = bias[gn];
#pragma unroll
            for (int r = 0; r < 8; ++r) {
                const int gm = m0 + wm * 32 + mi * 16 + r + hi * 8;
                float vv = acc[mi][ni][r] + bv;
                if (flags & 2) vv = fmaxf(vv, 0.f);
                if (flags & 4) vv += res[(size_t)gm * N + gn];
                if (flags & 1) ((u16b*)out)[(size_t)gm * N + gn] = f2bf(vv);
                else           ((float*)out)[(size_t)gm * N + gn] = vv;
            }
        }
    }
}

// ---------- fused causal ReLU-normalized attention ---------------------------
// q,k,v,ctx: bf16 (M,H) flat; head h lives at column offset h*D, row stride H.
// Block = 128 query rows of one (b,h); 8 waves x 16 rows; double-buffered
// 64-key K/V tiles staged with async global->LDS copies.
__global__ __launch_bounds__(256) void attn_kernel(
    const u16b* __restrict__ q, const u16b* __restrict__ k,
    const u16b* __restrict__ v, u16b* __restrict__ ctx)
{
    __shared__ __align__(16) u16b sQ[128 * 64];      // 16 KB
    __shared__ __align__(16) u16b sK[2][64 * 64];    // 16 KB
    __shared__ __align__(16) u16b sV[2][64 * 64];    // 16 KB
    __shared__ __align__(16) u16b sS[8][16 * 64];    // 16 KB

    const int tid  = threadIdx.x;
    const int wid  = tid >> 5;
    const int lane = tid & 31;
    const int hi   = lane >> 4;
    const int nl   = lane & 15;
    const int q0   = blockIdx.x * 128;
    const int bh   = blockIdx.y;
    const int b    = bh / NHEAD, h = bh % NHEAD;
    const size_t base = (size_t)b * SEQ * HID + (size_t)h * DHEAD;

    auto stage_kv = [&](int bi, int j0) {
#pragma unroll
        for (int i = 0; i < 2; ++i) {
            int flat = tid * 2 + i;
            int r = flat >> 3;
            int c = (flat & 7) << 3;
            cp16(&sK[bi][r * 64 + c], k + base + (size_t)(j0 + r) * HID + c);
            cp16(&sV[bi][r * 64 + c], v + base + (size_t)(j0 + r) * HID + c);
        }
    };

    // prologue: stage Q tile (128x64) + first K/V tile
#pragma unroll
    for (int i = 0; i < 4; ++i) {
        int flat = tid * 4 + i;
        int r = flat >> 3;
        int c = (flat & 7) << 3;
        cp16(&sQ[r * 64 + c], q + base + (size_t)(q0 + r) * HID + c);
    }
    stage_kv(0, 0);
    async_wait0();
    __syncthreads();

    v16bf qa[2];
    qa[0] = frag_nt(sQ, wid * 16, 64, 0);
    qa[1] = frag_nt(sQ, wid * 16, 64, 32);

    v8f acc[4];
#pragma unroll
    for (int nt = 0; nt < 4; ++nt) acc[nt] = zero8();
    float rs[8];
#pragma unroll
    for (int r = 0; r < 8; ++r) rs[r] = 0.f;

    const int kend = q0 + 128;               // causal truncation at tile level
    int buf = 0;
    for (int j0 = 0; j0 < kend; j0 += 64) {
        if (j0 + 64 < kend) stage_kv(buf ^ 1, j0 + 64);   // prefetch next tile

        // scores: 16q x 64k via 8 WMMAs, then scale/relu/causal, rowsum, restage
#pragma unroll
        for (int kt = 0; kt < 4; ++kt) {
            v8f s = zero8();
            s = wmma_bf16(qa[0], frag_nt(sK[buf], kt * 16, 64, 0),  s);
            s = wmma_bf16(qa[1], frag_nt(sK[buf], kt * 16, 64, 32), s);
            const int key = j0 + kt * 16 + nl;
#pragma unroll
            for (int r = 0; r < 8; ++r) {
                const int row = q0 + wid * 16 + r + hi * 8;
                float val = s[r] * 0.125f;                 // 1/sqrt(64)
                val = (key <= row) ? fmaxf(val, 0.f) : 0.f;
                rs[r] += val;
                sS[wid][(r + hi * 8) * 64 + kt * 16 + nl] = f2bf(val);
            }
        }
        __syncthreads();                      // sS restage visible (cross-lane)

        // ctx += att(16x64) @ V(64x64) : 8 WMMAs
#pragma unroll
        for (int kc = 0; kc < 2; ++kc) {
            v16bf aa = frag_nt(sS[wid], 0, 64, kc * 32);
#pragma unroll
            for (int nt = 0; nt < 4; ++nt)
                acc[nt] = wmma_bf16(aa, frag_nn(sV[buf], nt * 16, 64, kc * 32), acc[nt]);
        }

        async_wait0();                        // next K/V tile landed
        __syncthreads();                      // all waves done reading this buf
        buf ^= 1;
    }

    // row sums: reduce across the 16 lanes of each half (halves = m / m+8)
#pragma unroll
    for (int r = 0; r < 8; ++r) {
#pragma unroll
        for (int m = 1; m < 16; m <<= 1)
            rs[r] += __shfl_xor(rs[r], m, 32);
    }

    // normalize + store (bf16, flat (M,H) head layout)
#pragma unroll
    for (int nt = 0; nt < 4; ++nt) {
        const int gd = nt * 16 + nl;
#pragma unroll
        for (int r = 0; r < 8; ++r) {
            const int row = q0 + wid * 16 + r + hi * 8;
            float val = acc[nt][r] / (rs[r] + 1e-9f);
            ctx[base + (size_t)row * HID + gd] = f2bf(val);
        }
    }
}

// ---------- launch -----------------------------------------------------------

extern "C" void kernel_launch(void* const* d_in, const int* in_sizes, int n_in,
                              void* d_out, int out_size, void* d_ws, size_t ws_size,
                              hipStream_t stream) {
    (void)in_sizes; (void)n_in; (void)out_size; (void)ws_size;

    const float* x    = (const float*)d_in[0];
    const float* sfw  = (const float*)d_in[1];
    const float* sfb  = (const float*)d_in[2];
    const float* mask = (const float*)d_in[3];
    const float* wq   = (const float*)d_in[4];
    const float* bq   = (const float*)d_in[5];
    const float* wk   = (const float*)d_in[6];
    const float* bk   = (const float*)d_in[7];
    const float* wv   = (const float*)d_in[8];
    const float* bv   = (const float*)d_in[9];
    const float* wo   = (const float*)d_in[10];
    const float* bo   = (const float*)d_in[11];
    const float* ff1w = (const float*)d_in[12];
    const float* ff1b = (const float*)d_in[13];
    const float* ff2w = (const float*)d_in[14];
    const float* ff2b = (const float*)d_in[15];
    const float* g1   = (const float*)d_in[16];
    const float* b1   = (const float*)d_in[17];
    const float* g2   = (const float*)d_in[18];
    const float* b2   = (const float*)d_in[19];
    const float* g3   = (const float*)d_in[20];
    const float* b3   = (const float*)d_in[21];

    char* ws = (char*)d_ws;
    const size_t MBy = (size_t)1 << 20;
    float* x1  = (float*)(ws + 0 * MBy);     // 16 MB residual stream 1
    float* x2  = (float*)(ws + 16 * MBy);    // 16 MB residual stream 2
    u16b* lnb  = (u16b*)(ws + 32 * MBy);     //  8 MB LN output (reused 3x)
    u16b* qb   = (u16b*)(ws + 40 * MBy);     //  8 MB
    u16b* kb   = (u16b*)(ws + 48 * MBy);     //  8 MB
    u16b* vb   = (u16b*)(ws + 56 * MBy);     //  8 MB
    u16b* ctxb = (u16b*)(ws + 64 * MBy);     //  8 MB
    u16b* ffh  = (u16b*)(ws + 40 * MBy);     // 32 MB, aliases q/k/v/ctx (dead by then)
    u16b* sfwb = (u16b*)(ws + 72 * MBy);     //  2 MB
    u16b* wqb  = (u16b*)(ws + 74 * MBy);
    u16b* wkb  = (u16b*)(ws + 76 * MBy);
    u16b* wvb  = (u16b*)(ws + 78 * MBy);
    u16b* wob  = (u16b*)(ws + 80 * MBy);
    u16b* f1b  = (u16b*)(ws + 82 * MBy);     //  8 MB
    u16b* f2b  = (u16b*)(ws + 90 * MBy);     //  8 MB

    const int HH = HID * HID;                // 1M
    const int HF = HID * FFD;                // 4M
    cvt_mask_kernel<<<HH / 256, 256, 0, stream>>>(sfw, mask, sfwb, HH);
    cvt_kernel<<<HH / 256, 256, 0, stream>>>(wq, wqb, HH);
    cvt_kernel<<<HH / 256, 256, 0, stream>>>(wk, wkb, HH);
    cvt_kernel<<<HH / 256, 256, 0, stream>>>(wv, wvb, HH);
    cvt_kernel<<<HH / 256, 256, 0, stream>>>(wo, wob, HH);
    cvt_kernel<<<HF / 256, 256, 0, stream>>>(ff1w, f1b, HF);
    cvt_kernel<<<HF / 256, 256, 0, stream>>>(ff2w, f2b, HF);

    const dim3 blk(256);
    const dim3 gHH(HID / 128, MDIM / 128);   // (8, 32)

    // block 1: LN -> masked sparse linear -> +x
    ln_kernel<<<MDIM, blk, 0, stream>>>(x, g1, b1, lnb);
    gemm_bf16_kernel<<<gHH, blk, 0, stream>>>(lnb, sfwb, sfb, x, x1, MDIM, HID, HID, /*res*/4);

    // block 2: LN -> QKV -> relu-causal attention -> out proj -> +x1
    ln_kernel<<<MDIM, blk, 0, stream>>>(x1, g2, b2, lnb);
    gemm_bf16_kernel<<<gHH, blk, 0, stream>>>(lnb, wqb, bq, nullptr, qb, MDIM, HID, HID, /*bf16*/1);
    gemm_bf16_kernel<<<gHH, blk, 0, stream>>>(lnb, wkb, bk, nullptr, kb, MDIM, HID, HID, 1);
    gemm_bf16_kernel<<<gHH, blk, 0, stream>>>(lnb, wvb, bv, nullptr, vb, MDIM, HID, HID, 1);
    attn_kernel<<<dim3(SEQ / 128, BSZ * NHEAD), blk, 0, stream>>>(qb, kb, vb, ctxb);
    gemm_bf16_kernel<<<gHH, blk, 0, stream>>>(ctxb, wob, bo, x1, x2, MDIM, HID, HID, 4);

    // block 3: LN -> FFN -> +x2
    ln_kernel<<<MDIM, blk, 0, stream>>>(x2, g3, b3, lnb);
    gemm_bf16_kernel<<<dim3(FFD / 128, MDIM / 128), blk, 0, stream>>>(
        lnb, f1b, ff1b, nullptr, ffh, MDIM, FFD, HID, /*bf16|relu*/3);
    gemm_bf16_kernel<<<dim3(HID / 128, MDIM / 128), blk, 0, stream>>>(
        ffh, f2b, ff2b, x2, (float*)d_out, MDIM, HID, FFD, 4);
}
